// AttentiveBP_36910948942100
// MI455X (gfx1250) — matrix operand order repaired and checked by reading, not compiled, not run on hardware
//
#include <hip/hip_runtime.h>
#include <hip/hip_bf16.h>

// ---------------------------------------------------------------------------
// CDNA5 (gfx1250) WMMA types / helpers
// ---------------------------------------------------------------------------
typedef __attribute__((ext_vector_type(16))) _Float16 v16h;
typedef __attribute__((ext_vector_type(8)))  float    v8f;

struct Frag { union { v16h h; uint4 u[2]; }; };

__device__ inline v8f wmma_f16(const Frag& a, const Frag& b, v8f c) {
  // D = A(16x32 f16) x B(32x16 f16) + C(16x16 f32)
  return __builtin_amdgcn_wmma_f32_16x16x32_f16(false, a.h, false, b.h,
                                                (short)0, c, false, false);
}

// Branch-free fast transcendentals: v_exp_f32 + v_rcp_f32 only.
__device__ inline float fast_rcp(float x) { return __builtin_amdgcn_rcpf(x); }
__device__ inline float fast_exp(float x) {
  return __builtin_amdgcn_exp2f(x * 1.4426950408889634f);  // exp2(x*log2e)
}
__device__ inline float sigm(float x) { return fast_rcp(1.0f + fast_exp(-x)); }
__device__ inline float fast_tanh(float x) {
  return 1.0f - 2.0f * fast_rcp(fast_exp(2.0f * x) + 1.0f);
}

__device__ inline void atomicMaxF(float* p, float v) {
  if (v >= 0.0f) atomicMax((int*)p, __float_as_int(v));
  else           atomicMin((unsigned int*)p, __float_as_uint(v));
}

// ---------------------------------------------------------------------------
// Fused 2-layer GRU over 16-edge tiles; hidden state register/LDS resident
// across all T timesteps. Layer0 input dim == 1 (scalar path); recurrent
// matmuls + layer1 input matmul via v_wmma_f32_16x16x32_f16.
// ---------------------------------------------------------------------------
#define T_STEPS 10

__global__ __launch_bounds__(128)
void gru_wmma_kernel(const float* __restrict__ msg,      // [E][T]
                     const float* __restrict__ h0in,     // [2][E][32]
                     const _Float16* __restrict__ whh0f, // [96][32] f16
                     const _Float16* __restrict__ wih1f, // [96][32] f16
                     const _Float16* __restrict__ whh1f, // [96][32] f16
                     const float* __restrict__ wih0,     // [96]
                     const float* __restrict__ bih0, const float* __restrict__ bhh0,
                     const float* __restrict__ bih1, const float* __restrict__ bhh1,
                     float* __restrict__ hout,           // [2][E][32]
                     int E) {
  __shared__ _Float16 ldsAll[4][2][16 * 32];
  const int wave = threadIdx.x >> 5;
  const int lane = threadIdx.x & 31;
  const int tile = blockIdx.x * 4 + wave;
  const int ntiles = E >> 4;
  if (tile >= ntiles) return;          // wave-uniform; EXEC stays all-ones
  const int half = lane >> 4;
  const int q = lane & 15;
  const int base = tile << 4;
  _Float16* lds0 = &ldsAll[wave][0][0];
  _Float16* lds1 = &ldsAll[wave][1][0];

  // B fragments (K=32): lane n=q holds Bt[n][k] in table order (k=half*8+0..7, 16+half*8+0..7); Bt == Whh/Wih as-is.
  Frag fwhh0[6], fwih1[6], fwhh1[6];
#pragma unroll
  for (int c = 0; c < 6; ++c) {
    const int n = c * 16 + q;
    const _Float16* p0 = whh0f + n * 32 + half * 8;
    fwhh0[c].u[0] = *(const uint4*)p0; fwhh0[c].u[1] = *(const uint4*)(p0 + 16);
    const _Float16* p1 = wih1f + n * 32 + half * 8;
    fwih1[c].u[0] = *(const uint4*)p1; fwih1[c].u[1] = *(const uint4*)(p1 + 16);
    const _Float16* p2 = whh1f + n * 32 + half * 8;
    fwhh1[c].u[0] = *(const uint4*)p2; fwhh1[c].u[1] = *(const uint4*)(p2 + 16);
  }

  // Per-lane gate constants (columns j = cc*16+q of the C layout).
  float wr0[2], wz0[2], wn0[2], br0[2], bz0[2], bn0i[2], bn0h[2];
  float br1[2], bz1[2], bn1i[2], bn1h[2];
#pragma unroll
  for (int cc = 0; cc < 2; ++cc) {
    const int j = cc * 16 + q;
    wr0[cc] = wih0[j]; wz0[cc] = wih0[32 + j]; wn0[cc] = wih0[64 + j];
    br0[cc] = bih0[j] + bhh0[j];
    bz0[cc] = bih0[32 + j] + bhh0[32 + j];
    bn0i[cc] = bih0[64 + j]; bn0h[cc] = bhh0[64 + j];
    br1[cc] = bih1[j] + bhh1[j];
    bz1[cc] = bih1[32 + j] + bhh1[32 + j];
    bn1i[cc] = bih1[64 + j]; bn1h[cc] = bhh1[64 + j];
  }

  // Initial hidden states (C layout: lane col = q, rows v + half*8).
  float h0s[2][8], h1s[2][8];
#pragma unroll
  for (int cc = 0; cc < 2; ++cc)
#pragma unroll
    for (int v = 0; v < 8; ++v) {
      const int row = v + half * 8;
      const int j = cc * 16 + q;
      const float a = h0in[(size_t)(base + row) * 32 + j];
      const float b = h0in[(size_t)E * 32 + (size_t)(base + row) * 32 + j];
      h0s[cc][v] = a; h1s[cc][v] = b;
      lds0[row * 32 + j] = (_Float16)a;
      lds1[row * 32 + j] = (_Float16)b;
    }
  asm volatile("s_wait_dscnt 0" ::: "memory");

  for (int t = 0; t < T_STEPS; ++t) {
    // ---- layer 0: gh = h0 @ Whh0^T  (6 WMMAs) ----
    Frag a0;
    { const _Float16* p = lds0 + q * 32 + half * 8;
      a0.u[0] = *(const uint4*)p; a0.u[1] = *(const uint4*)(p + 16); }
    v8f acc[6];
#pragma unroll
    for (int c = 0; c < 6; ++c) { v8f z = {}; acc[c] = wmma_f16(a0, fwhh0[c], z); }

    float mv[8];
#pragma unroll
    for (int v = 0; v < 8; ++v)
      mv[v] = msg[(size_t)(base + v + half * 8) * T_STEPS + t];

#pragma unroll
    for (int cc = 0; cc < 2; ++cc)
#pragma unroll
      for (int v = 0; v < 8; ++v) {
        const float m = mv[v];
        const float r = sigm(m * wr0[cc] + br0[cc] + acc[cc][v]);
        const float z = sigm(m * wz0[cc] + bz0[cc] + acc[2 + cc][v]);
        const float n = fast_tanh(m * wn0[cc] + bn0i[cc] + r * (acc[4 + cc][v] + bn0h[cc]));
        h0s[cc][v] = (1.0f - z) * n + z * h0s[cc][v];
      }
#pragma unroll
    for (int cc = 0; cc < 2; ++cc)
#pragma unroll
      for (int v = 0; v < 8; ++v)
        lds0[(v + half * 8) * 32 + cc * 16 + q] = (_Float16)h0s[cc][v];
    asm volatile("s_wait_dscnt 0" ::: "memory");

    // ---- layer 1: gi = x@Wih1^T, gh = h1@Whh1^T (r,z chained; n split) ----
    Frag ax, a1;
    { const _Float16* p = lds0 + q * 32 + half * 8;
      ax.u[0] = *(const uint4*)p; ax.u[1] = *(const uint4*)(p + 16); }
    { const _Float16* p = lds1 + q * 32 + half * 8;
      a1.u[0] = *(const uint4*)p; a1.u[1] = *(const uint4*)(p + 16); }
    v8f arz[4], gin[2], ghn[2];
#pragma unroll
    for (int c = 0; c < 4; ++c) {
      v8f z = {};
      arz[c] = wmma_f16(ax, fwih1[c], z);
      arz[c] = wmma_f16(a1, fwhh1[c], arz[c]);
    }
#pragma unroll
    for (int c = 0; c < 2; ++c) {
      v8f z = {}; gin[c] = wmma_f16(ax, fwih1[4 + c], z);
      v8f z2 = {}; ghn[c] = wmma_f16(a1, fwhh1[4 + c], z2);
    }
#pragma unroll
    for (int cc = 0; cc < 2; ++cc)
#pragma unroll
      for (int v = 0; v < 8; ++v) {
        const float r = sigm(arz[cc][v] + br1[cc]);
        const float z = sigm(arz[2 + cc][v] + bz1[cc]);
        const float n = fast_tanh(gin[cc][v] + bn1i[cc] + r * (ghn[cc][v] + bn1h[cc]));
        h1s[cc][v] = (1.0f - z) * n + z * h1s[cc][v];
      }
#pragma unroll
    for (int cc = 0; cc < 2; ++cc)
#pragma unroll
      for (int v = 0; v < 8; ++v)
        lds1[(v + half * 8) * 32 + cc * 16 + q] = (_Float16)h1s[cc][v];
    asm volatile("s_wait_dscnt 0" ::: "memory");
  }

#pragma unroll
  for (int cc = 0; cc < 2; ++cc)
#pragma unroll
    for (int v = 0; v < 8; ++v) {
      const int row = v + half * 8;
      const int j = cc * 16 + q;
      hout[(size_t)(base + row) * 32 + j] = h0s[cc][v];
      hout[(size_t)E * 32 + (size_t)(base + row) * 32 + j] = h1s[cc][v];
    }
}

// ---------------------------------------------------------------------------
// Generic WMMA GEMM: C[M][N] = A[M][K](f16) * Bt[N][K](f16)^T (+bias)
// One wave per 16x16 tile; K % 32 == 0; M,N % 16 == 0.
// ---------------------------------------------------------------------------
__global__ __launch_bounds__(128)
void gemm_wmma_kernel(const _Float16* __restrict__ A,
                      const _Float16* __restrict__ Bt,
                      const float* __restrict__ bias,
                      float* __restrict__ C,
                      int M, int N, int K) {
  const int lane = threadIdx.x & 31;
  const int wave = threadIdx.x >> 5;
  const int tm = blockIdx.x * 4 + wave;
  const int tn = blockIdx.y;
  if (tm * 16 >= M) return;
  const int half = lane >> 4;
  const int q = lane & 15;
  v8f acc = {};
  for (int k0 = 0; k0 < K; k0 += 32) {
    Frag a, b;
    const _Float16* pa = A + (size_t)(tm * 16 + q) * K + k0 + half * 8;
    a.u[0] = *(const uint4*)pa; a.u[1] = *(const uint4*)(pa + 16);
    const _Float16* pb = Bt + (size_t)(tn * 16 + q) * K + k0 + half * 8;
    b.u[0] = *(const uint4*)pb; b.u[1] = *(const uint4*)(pb + 16);
    acc = wmma_f16(a, b, acc);
  }
  const float bv = bias ? bias[tn * 16 + q] : 0.0f;
#pragma unroll
  for (int v = 0; v < 8; ++v) {
    const int row = tm * 16 + v + half * 8;
    C[(size_t)row * N + tn * 16 + q] = acc[v] + bv;
  }
}

// ---------------------------------------------------------------------------
// Prep / elementwise kernels
// ---------------------------------------------------------------------------
__global__ void cvt_f16_kernel(_Float16* dst, const float* src, int n) {
  int i = blockIdx.x * blockDim.x + threadIdx.x;
  if (i < n) dst[i] = (_Float16)src[i];
}
__global__ void cvt_f16_pad_kernel(_Float16* dst, const float* src,
                                   int rows, int cols, int dcols) {
  int i = blockIdx.x * blockDim.x + threadIdx.x;
  if (i >= rows * dcols) return;
  int r = i / dcols, c = i % dcols;
  dst[i] = (c < cols) ? (_Float16)src[r * cols + c] : (_Float16)0.0f;
}
__global__ void transpose_cvt_kernel(_Float16* dst, const float* src, int K, int N) {
  // src [K][N] f32 -> dst [N][K] f16
  int i = blockIdx.x * blockDim.x + threadIdx.x;
  if (i >= K * N) return;
  int n = i / K, k = i % K;
  dst[(size_t)n * K + k] = (_Float16)src[(size_t)k * N + n];
}
__global__ void fill_kernel(float* p, float v, int n) {
  int i = blockIdx.x * blockDim.x + threadIdx.x;
  if (i < n) p[i] = v;
}
__global__ void build_ea1_kernel(_Float16* dst, const float* lc,
                                 const float* hid, int E) {
  int i = blockIdx.x * blockDim.x + threadIdx.x;   // over E*96
  if (i >= E * 96) return;
  int e = i / 96, c = i % 96;
  float v;
  if (c == 0) v = lc[e];
  else if (c < 65) { int cc = c - 1; int l = cc >> 5; int j = cc & 31;
                     v = hid[((size_t)l * E + e) * 32 + j]; }
  else v = 0.0f;
  dst[i] = (_Float16)v;
}
__global__ void build_ea2_kernel(_Float16* dst, const float* hid, int E) {
  int i = blockIdx.x * blockDim.x + threadIdx.x;   // over E*64
  if (i >= E * 64) return;
  int e = i / 64, cc = i % 64; int l = cc >> 5, j = cc & 31;
  dst[i] = (_Float16)hid[((size_t)l * E + e) * 32 + j];
}
__global__ void ea_mean_kernel(float* mean, const float* ea, int E) {
  int col = threadIdx.x & 31;
  int rw = blockIdx.x * (blockDim.x >> 5) + (threadIdx.x >> 5);
  int stride = gridDim.x * (blockDim.x >> 5);
  float s = 0.0f;
  for (int e = rw; e < E; e += stride) s += ea[(size_t)e * 32 + col];
  atomicAdd(&mean[col], s);
}
__global__ void scale_kernel(float* p, float s, int n) {
  int i = blockIdx.x * blockDim.x + threadIdx.x;
  if (i < n) p[i] *= s;
}

// ---------------------------------------------------------------------------
// GAT kernels
// ---------------------------------------------------------------------------
__global__ void compute_ve_kernel(float* ve, const float* We,
                                  const float* att_e, int cout) {
  int i = blockIdx.x * blockDim.x + threadIdx.x;   // 4*32
  if (i >= 128) return;
  int hd = i >> 5, r = i & 31;
  int hc = 4 * cout;
  float s = 0.0f;
  for (int c = 0; c < cout; ++c) s += We[(size_t)r * hc + hd * cout + c] * att_e[hd * cout + c];
  ve[hd * 32 + r] = s;
}
__global__ void gat_asad_kernel(float* a_s, float* a_d, const float* h,
                                const float* att_s, const float* att_d,
                                int N, int cout) {
  int n = blockIdx.x * blockDim.x + threadIdx.x;
  if (n >= N) return;
  int hc = 4 * cout;
  for (int hd = 0; hd < 4; ++hd) {
    float ss = 0.0f, sd = 0.0f;
    for (int c = 0; c < cout; ++c) {
      float hv = h[(size_t)n * hc + hd * cout + c];
      ss += hv * att_s[hd * cout + c];
      sd += hv * att_d[hd * cout + c];
    }
    a_s[n * 4 + hd] = ss; a_d[n * 4 + hd] = sd;
  }
}
__global__ void gat_edge_alpha_kernel(const int* __restrict__ ei,
                                      const float* __restrict__ ea,
                                      const float* __restrict__ mean_ea,
                                      const float* __restrict__ ve,
                                      const float* __restrict__ a_s,
                                      const float* __restrict__ a_d,
                                      float* __restrict__ alpha,
                                      float* __restrict__ amax,
                                      int Etot, int Nn) {
  int e = blockIdx.x * blockDim.x + threadIdx.x;
  if (e >= Etot + Nn) return;
  int s_, d_;
  const float* er;
  if (e < Etot) { s_ = ei[e]; d_ = ei[Etot + e]; er = ea + (size_t)e * 32; }
  else          { s_ = d_ = e - Etot; er = mean_ea; }
  for (int hd = 0; hd < 4; ++hd) {
    float dotv = 0.0f;
    for (int i = 0; i < 32; ++i) dotv += er[i] * ve[hd * 32 + i];
    float xv = a_s[s_ * 4 + hd] + a_d[d_ * 4 + hd] + dotv;
    xv = xv > 0.0f ? xv : 0.2f * xv;
    alpha[(size_t)e * 4 + hd] = xv;
    atomicMaxF(&amax[d_ * 4 + hd], xv);
  }
}
__global__ void gat_edge_exp_kernel(const int* __restrict__ ei,
                                    float* __restrict__ alpha,
                                    const float* __restrict__ amax,
                                    float* __restrict__ den,
                                    int Etot, int Nn) {
  int e = blockIdx.x * blockDim.x + threadIdx.x;
  if (e >= Etot + Nn) return;
  int d_ = (e < Etot) ? ei[Etot + e] : (e - Etot);
  for (int hd = 0; hd < 4; ++hd) {
    float ex = fast_exp(alpha[(size_t)e * 4 + hd] - amax[d_ * 4 + hd]);
    alpha[(size_t)e * 4 + hd] = ex;
    atomicAdd(&den[d_ * 4 + hd], ex);
  }
}
__global__ void gat_agg_kernel(const int* __restrict__ ei,
                               const float* __restrict__ ex,
                               const float* __restrict__ den,
                               const float* __restrict__ h,
                               float* __restrict__ agg,
                               int Etot, int Nn, int cout) {
  int e = blockIdx.x * blockDim.x + threadIdx.x;
  if (e >= Etot + Nn) return;
  int s_, d_;
  if (e < Etot) { s_ = ei[e]; d_ = ei[Etot + e]; } else { s_ = d_ = e - Etot; }
  int hc = 4 * cout;
  for (int hd = 0; hd < 4; ++hd) {
    float w = ex[(size_t)e * 4 + hd] * fast_rcp(den[d_ * 4 + hd] + 1e-16f);
    for (int c = 0; c < cout; ++c)
      atomicAdd(&agg[(size_t)d_ * hc + hd * cout + c],
                h[(size_t)s_ * hc + hd * cout + c] * w);
  }
}
__global__ void gat_fin_concat_kernel(_Float16* xnext, const float* agg,
                                      const float* bias, int N, int hc) {
  int i = blockIdx.x * blockDim.x + threadIdx.x;
  if (i >= N * hc) return;
  float v = agg[i] + bias[i % hc];
  v = v > 0.0f ? v : 0.01f * v;
  xnext[i] = (_Float16)v;
}
__global__ void gat_fin_mean_kernel(float* hfin, const float* agg,
                                    const float* bias, int N) {
  int i = blockIdx.x * blockDim.x + threadIdx.x;   // N*64
  if (i >= N * 64) return;
  int n = i / 64, c = i % 64;
  float v = 0.0f;
  for (int hd = 0; hd < 4; ++hd) v += agg[(size_t)n * 256 + hd * 64 + c];
  v = v * 0.25f + bias[c];
  v = v > 0.0f ? v : 0.01f * v;
  hfin[i] = v;
}

// ---------------------------------------------------------------------------
// Pooling + folded attention + ragged segment softmax
// ---------------------------------------------------------------------------
__global__ void pool_scatter_kernel(float* pool, const float* hfin,
                                    const int* idx, int N) {
  int i = blockIdx.x * blockDim.x + threadIdx.x;   // N*64
  if (i >= N * 64) return;
  int n = i / 64, c = i % 64;
  atomicAdd(&pool[(size_t)idx[n] * 64 + c], hfin[i]);
}
__global__ void pool_div_kernel(float* out, const float* pool,
                                const float* dom, int D) {
  int i = blockIdx.x * blockDim.x + threadIdx.x;   // D*64
  if (i >= D * 64) return;
  int d = i / 64;
  out[i] = pool[(size_t)(d + 1) * 64 + (i % 64)] * fast_rcp(dom[d]);
}
__global__ void attn_fold_kernel(float* vq, float* vk, const float* Wq,
                                 const float* Wk, const float* Ws) {
  int i = blockIdx.x * blockDim.x + threadIdx.x;   // 4*64
  if (i >= 256) return;
  int hd = i >> 6, ii = i & 63;
  float sq = 0.0f, sk = 0.0f;
  for (int o = 0; o < 64; ++o) {
    sq += Wq[(size_t)hd * 4096 + o * 64 + ii] * Ws[hd * 128 + o];
    sk += Wk[(size_t)hd * 4096 + o * 64 + ii] * Ws[hd * 128 + 64 + o];
  }
  vq[i] = sq; vk[i] = sk;
}
__global__ void aqak_kernel(float* aq, float* ak, const float* pool,
                            const float* vq, const float* vk, int D) {
  int i = blockIdx.x * blockDim.x + threadIdx.x;   // D*4
  if (i >= D * 4) return;
  int d = i >> 2, hd = i & 3;
  float sq = 0.0f, sk = 0.0f;
  for (int c = 0; c < 64; ++c) {
    float pv = pool[(size_t)d * 64 + c];
    sq += pv * vq[hd * 64 + c];
    sk += pv * vk[hd * 64 + c];
  }
  aq[i] = sq; ak[i] = sk;
}
__global__ void nbr_score_max_kernel(float* sbuf, float* segm,
                                     const int* nbr_src, const int* nbr_seg,
                                     const float* aq, const float* ak,
                                     const float* bs, int G, int D) {
  int g = blockIdx.x * blockDim.x + threadIdx.x;
  if (g >= G) return;
  int idx = nbr_src[g];
  int i = idx / D, j = idx % D;
  int seg = nbr_seg[g];
  for (int hd = 0; hd < 4; ++hd) {
    float s = sigm(aq[i * 4 + hd] + ak[j * 4 + hd] + bs[hd]);
    sbuf[(size_t)g * 4 + hd] = s;
    // sigmoid output > 0 so int-punned max is order-preserving (init 0)
    atomicMax((int*)&segm[(size_t)seg * 4 + hd], __float_as_int(s));
  }
}
__global__ void nbr_exp_kernel(float* sbuf, float* segden, const int* nbr_seg,
                               const float* segm, int G) {
  int g = blockIdx.x * blockDim.x + threadIdx.x;
  if (g >= G) return;
  int seg = nbr_seg[g];
  for (int hd = 0; hd < 4; ++hd) {
    float ex = fast_exp(sbuf[(size_t)g * 4 + hd] - segm[(size_t)seg * 4 + hd]);
    sbuf[(size_t)g * 4 + hd] = ex;
    atomicAdd(&segden[(size_t)seg * 4 + hd], ex);
  }
}
__global__ void nbr_norm_kernel(float* out, const float* sbuf,
                                const float* segden, const int* nbr_seg,
                                const int* nbr_scale, int G) {
  int g = blockIdx.x * blockDim.x + threadIdx.x;
  if (g >= G) return;
  int seg = nbr_seg[g];
  float sc = (float)nbr_scale[g];
  for (int hd = 0; hd < 4; ++hd)
    out[(size_t)g * 4 + hd] =
        sbuf[(size_t)g * 4 + hd] * fast_rcp(segden[(size_t)seg * 4 + hd]) * sc;
}

// ---------------------------------------------------------------------------
// Host orchestration
// ---------------------------------------------------------------------------
extern "C" void kernel_launch(void* const* d_in, const int* in_sizes, int n_in,
                              void* d_out, int out_size, void* d_ws, size_t ws_size,
                              hipStream_t stream) {
  (void)n_in; (void)out_size; (void)ws_size;

  const float* x        = (const float*)d_in[0];
  const int*   ei       = (const int*)d_in[1];
  const float* lc       = (const float*)d_in[2];
  const float* msg1     = (const float*)d_in[3];
  const float* h0g1     = (const float*)d_in[4];
  const float* msg2     = (const float*)d_in[5];
  const float* h0g2     = (const float*)d_in[6];
  const int*   scidx    = (const int*)d_in[7];
  const float* scdom    = (const float*)d_in[8];
  const int*   nbrsrc   = (const int*)d_in[9];
  const int*   nbrseg   = (const int*)d_in[10];
  const int*   nbrscale = (const int*)d_in[11];
  // d_in[12] = num_groups (device scalar; NSEG fixed by problem spec)

  const int N    = in_sizes[0] / 64;
  const int Etot = in_sizes[1] / 2;
  const int E1   = in_sizes[2];
  const int E2   = Etot - E1;
  const int D    = in_sizes[8];
  const int G    = in_sizes[9];
  const int NSEG = 65536;

  const float* g1p[8]; for (int i = 0; i < 8; ++i) g1p[i] = (const float*)d_in[13 + i];
  const float* g2p[8]; for (int i = 0; i < 8; ++i) g2p[i] = (const float*)d_in[21 + i];
  const float* proj1W = (const float*)d_in[29]; const float* proj1b = (const float*)d_in[30];
  const float* proj2W = (const float*)d_in[31]; const float* proj2b = (const float*)d_in[32];
  struct GatH { const float *W, *as, *ad, *We, *ae, *b; int cin, cout; } gat[4];
  for (int l = 0; l < 4; ++l) {
    int o = 33 + l * 6;
    gat[l] = { (const float*)d_in[o], (const float*)d_in[o + 1], (const float*)d_in[o + 2],
               (const float*)d_in[o + 3], (const float*)d_in[o + 4], (const float*)d_in[o + 5],
               (l == 0) ? 64 : 32, (l == 3) ? 64 : 8 };
  }
  const float* Wq = (const float*)d_in[57]; const float* Wk = (const float*)d_in[58];
  const float* Ws = (const float*)d_in[59]; const float* bs = (const float*)d_in[60];

  float* outW = (float*)d_out;                 // [G][4]
  float* hid1 = outW + (size_t)G * 4;          // [2][E1][32]
  float* hid2 = hid1 + (size_t)2 * E1 * 32;    // [2][E2][32]

  // ---- workspace carving (256B aligned) ----
  unsigned char* wsb = (unsigned char*)d_ws;
  size_t off = 0;
  auto carve = [&](size_t bytes) -> void* {
    void* p = wsb + off;
    off += (bytes + 255) & ~(size_t)255;
    return p;
  };
  _Float16* g1whh0 = (_Float16*)carve(96 * 32 * 2);
  _Float16* g1wih1 = (_Float16*)carve(96 * 32 * 2);
  _Float16* g1whh1 = (_Float16*)carve(96 * 32 * 2);
  _Float16* g2whh0 = (_Float16*)carve(96 * 32 * 2);
  _Float16* g2wih1 = (_Float16*)carve(96 * 32 * 2);
  _Float16* g2whh1 = (_Float16*)carve(96 * 32 * 2);
  _Float16* proj1t = (_Float16*)carve(32 * 96 * 2);
  _Float16* proj2t = (_Float16*)carve(32 * 64 * 2);
  _Float16* wt[4];
  for (int l = 0; l < 4; ++l)
    wt[l] = (_Float16*)carve((size_t)(4 * gat[l].cout) * gat[l].cin * 2);
  float* veL    = (float*)carve(4 * 32 * 4);
  float* meanea = (float*)carve(32 * 4);
  // regionA: ea-input f16 (early)  <->  GAT gemm output hbuf (later)
  size_t rA = (size_t)E1 * 96 * 2; size_t rA2 = (size_t)N * 256 * 4;
  void* regA = carve(rA > rA2 ? rA : rA2);
  _Float16* eain = (_Float16*)regA;
  float*    hbuf = (float*)regA;
  float* eaBuf = (float*)carve((size_t)Etot * 32 * 4);
  _Float16* xf16 = (_Float16*)carve((size_t)N * 64 * 2);
  float* agg  = (float*)carve((size_t)N * 256 * 4);
  float* a_s  = (float*)carve((size_t)N * 4 * 4);
  float* a_d  = (float*)carve((size_t)N * 4 * 4);
  float* amax = (float*)carve((size_t)N * 4 * 4);
  float* den  = (float*)carve((size_t)N * 4 * 4);
  // regionB: edge alpha/ex (GAT phase) <-> score buffer (final phase)
  size_t rB = (size_t)(Etot + N) * 4 * 4; size_t rB2 = (size_t)G * 4 * 4;
  void* regB = carve(rB > rB2 ? rB : rB2);
  float* alphaB = (float*)regB;
  float* sbuf   = (float*)regB;
  float* hfin   = (float*)carve((size_t)N * 64 * 4);
  float* pool   = (float*)carve((size_t)(D + 1) * 64 * 4);
  float* poolf  = (float*)carve((size_t)D * 64 * 4);
  float* vq     = (float*)carve(256 * 4);
  float* vk     = (float*)carve(256 * 4);
  float* aq     = (float*)carve((size_t)D * 4 * 4);
  float* ak     = (float*)carve((size_t)D * 4 * 4);
  float* segm   = (float*)carve((size_t)NSEG * 4 * 4);
  float* segden = (float*)carve((size_t)NSEG * 4 * 4);

  auto nb = [](long n, int b) { return (unsigned)((n + b - 1) / b); };

  // ---- 1. weight prep (f32 -> f16, fragment-friendly layouts) ----
  cvt_f16_kernel<<<nb(96 * 32, 256), 256, 0, stream>>>(g1whh0, g1p[1], 96 * 32);
  cvt_f16_kernel<<<nb(96 * 32, 256), 256, 0, stream>>>(g1wih1, g1p[4], 96 * 32);
  cvt_f16_kernel<<<nb(96 * 32, 256), 256, 0, stream>>>(g1whh1, g1p[5], 96 * 32);
  cvt_f16_kernel<<<nb(96 * 32, 256), 256, 0, stream>>>(g2whh0, g2p[1], 96 * 32);
  cvt_f16_kernel<<<nb(96 * 32, 256), 256, 0, stream>>>(g2wih1, g2p[4], 96 * 32);
  cvt_f16_kernel<<<nb(96 * 32, 256), 256, 0, stream>>>(g2whh1, g2p[5], 96 * 32);
  cvt_f16_pad_kernel<<<nb(32 * 96, 256), 256, 0, stream>>>(proj1t, proj1W, 32, 65, 96);
  cvt_f16_kernel<<<nb(32 * 64, 256), 256, 0, stream>>>(proj2t, proj2W, 32 * 64);
  for (int l = 0; l < 4; ++l) {
    int hc = 4 * gat[l].cout;
    transpose_cvt_kernel<<<nb((long)gat[l].cin * hc, 256), 256, 0, stream>>>(
        wt[l], gat[l].W, gat[l].cin, hc);
  }

  // ---- 2. GRUs (WMMA core) -> hidden directly into d_out ----
  {
    int ntiles = E1 >> 4;
    gru_wmma_kernel<<<nb(ntiles, 4), 128, 0, stream>>>(
        msg1, h0g1, g1whh0, g1wih1, g1whh1,
        g1p[0], g1p[2], g1p[3], g1p[6], g1p[7], hid1, E1);
  }
  {
    int ntiles = E2 >> 4;
    gru_wmma_kernel<<<nb(ntiles, 4), 128, 0, stream>>>(
        msg2, h0g2, g2whh0, g2wih1, g2whh1,
        g2p[0], g2p[2], g2p[3], g2p[6], g2p[7], hid2, E2);
  }

  // ---- 3. edge-attr projections (WMMA gemm) ----
  build_ea1_kernel<<<nb((long)E1 * 96, 256), 256, 0, stream>>>(eain, lc, hid1, E1);
  gemm_wmma_kernel<<<dim3(nb(E1 / 16, 4), 2), 128, 0, stream>>>(
      eain, proj1t, proj1b, eaBuf, E1, 32, 96);
  build_ea2_kernel<<<nb((long)E2 * 64, 256), 256, 0, stream>>>(eain, hid2, E2);
  gemm_wmma_kernel<<<dim3(nb(E2 / 16, 4), 2), 128, 0, stream>>>(
      eain, proj2t, proj2b, eaBuf + (size_t)E1 * 32, E2, 32, 64);

  // mean edge-attr (self-loop fill_value='mean')
  fill_kernel<<<1, 32, 0, stream>>>(meanea, 0.0f, 32);
  ea_mean_kernel<<<128, 256, 0, stream>>>(meanea, eaBuf, Etot);
  scale_kernel<<<1, 32, 0, stream>>>(meanea, 1.0f / (float)Etot, 32);

  // ---- 4. GAT stack ----
  cvt_f16_kernel<<<nb((long)N * 64, 256), 256, 0, stream>>>(xf16, x, N * 64);
  const int EN = Etot + N;
  for (int l = 0; l < 4; ++l) {
    const int cout = gat[l].cout, hc = 4 * cout, cin = gat[l].cin;
    gemm_wmma_kernel<<<dim3(nb(N / 16, 4), hc / 16), 128, 0, stream>>>(
        xf16, wt[l], nullptr, hbuf, N, hc, cin);
    compute_ve_kernel<<<1, 128, 0, stream>>>(veL, gat[l].We, gat[l].ae, cout);
    gat_asad_kernel<<<nb(N, 256), 256, 0, stream>>>(a_s, a_d, hbuf,
                                                    gat[l].as, gat[l].ad, N, cout);
    fill_kernel<<<nb((long)N * 4, 256), 256, 0, stream>>>(amax, -1e30f, N * 4);
    fill_kernel<<<nb((long)N * 4, 256), 256, 0, stream>>>(den, 0.0f, N * 4);
    fill_kernel<<<nb((long)N * hc, 256), 256, 0, stream>>>(agg, 0.0f, N * hc);
    gat_edge_alpha_kernel<<<nb(EN, 256), 256, 0, stream>>>(
        ei, eaBuf, meanea, veL, a_s, a_d, alphaB, amax, Etot, N);
    gat_edge_exp_kernel<<<nb(EN, 256), 256, 0, stream>>>(ei, alphaB, amax, den, Etot, N);
    gat_agg_kernel<<<nb(EN, 256), 256, 0, stream>>>(ei, alphaB, den, hbuf, agg,
                                                    Etot, N, cout);
    if (l < 3)
      gat_fin_concat_kernel<<<nb((long)N * hc, 256), 256, 0, stream>>>(
          xf16, agg, gat[l].b, N, hc);
    else
      gat_fin_mean_kernel<<<nb((long)N * 64, 256), 256, 0, stream>>>(
          hfin, agg, gat[l].b, N);
  }

  // ---- 5. pooling ----
  fill_kernel<<<nb((long)(D + 1) * 64, 256), 256, 0, stream>>>(pool, 0.0f, (D + 1) * 64);
  pool_scatter_kernel<<<nb((long)N * 64, 256), 256, 0, stream>>>(pool, hfin, scidx, N);
  pool_div_kernel<<<nb((long)D * 64, 256), 256, 0, stream>>>(poolf, pool, scdom, D);

  // ---- 6. folded attention + ragged per-group softmax ----
  attn_fold_kernel<<<1, 256, 0, stream>>>(vq, vk, Wq, Wk, Ws);
  aqak_kernel<<<nb((long)D * 4, 256), 256, 0, stream>>>(aq, ak, poolf, vq, vk, D);
  fill_kernel<<<nb((long)NSEG * 4, 256), 256, 0, stream>>>(segm, 0.0f, NSEG * 4);
  fill_kernel<<<nb((long)NSEG * 4, 256), 256, 0, stream>>>(segden, 0.0f, NSEG * 4);
  nbr_score_max_kernel<<<nb(G, 256), 256, 0, stream>>>(sbuf, segm, nbrsrc, nbrseg,
                                                       aq, ak, bs, G, D);
  nbr_exp_kernel<<<nb(G, 256), 256, 0, stream>>>(sbuf, segden, nbrseg, segm, G);
  nbr_norm_kernel<<<nb(G, 256), 256, 0, stream>>>(outW, sbuf, segden, nbrseg,
                                                  nbrscale, G);
}
